// BurgersApproximator_12919261626779
// MI455X (gfx1250) — matrix operand order repaired and checked by reading, not compiled, not run on hardware
//
#include <hip/hip_runtime.h>

typedef __attribute__((ext_vector_type(16))) _Float16 v16h;
typedef __attribute__((ext_vector_type(8)))  float    v8f;

#define WMMA_F16(a, b, c) \
  __builtin_amdgcn_wmma_f32_16x16x32_f16(false, (a), false, (b), (short)0, (c), false, false)

struct __align__(8) H4 { _Float16 v[4]; };

// ---- fragment index math (CDNA5 16-bit WMMA layouts, CDNA5 ISA) ----
// A (16x32 f16): lanes 0-15 row M=lane; halves h -> K: (h<8 ? h : h+8), lanes 16-31 add 8.
__device__ __forceinline__ int kmapA(int h, int lane) {
  int k = (h < 8) ? h : (h + 8);
  return k + ((lane & 16) ? 8 : 0);
}
// B (32x16 f16): same K order as A: halves h -> K: (h<8 ? h : h+8), lanes 16-31 add 8.
__device__ __forceinline__ int kmapB(int h, int lane) {
  return ((h < 8) ? h : (h + 8)) + ((lane & 16) ? 8 : 0);
}

__device__ __forceinline__ v16h load_fragA_lds(const _Float16* s, int lane, int kbase, int ld) {
  const int m = lane & 15;
  v16h a;
#pragma unroll
  for (int h = 0; h < 16; ++h) a[h] = s[m * ld + kbase + kmapA(h, lane)];
  return a;
}

__device__ __forceinline__ v16h load_fragB(const _Float16* p, int lane) {
  // packed layout: fragment = 32 lanes x 16 contiguous halves (32B per lane)
  return *(const v16h*)(p + (size_t)lane * 16);
}

// ---- generic 2-layer MLP on one wave, MT row-tiles of 16:
//      [MT*16 x KC*32] -> relu(128) -> 64 (+bias, no relu). B fragments reused across MT.
template <int KC, int MT>
__device__ __forceinline__ void mlp2(const _Float16* sIn, _Float16* sH,
                                     const _Float16* pW1, const float* b1,
                                     const _Float16* pW2, const float* b2,
                                     v8f (&acc2)[MT][4], int lane) {
  const int nl = lane & 15;
  const int mh = (lane & 16) ? 8 : 0;
  v16h A[MT][KC];
#pragma unroll
  for (int mt = 0; mt < MT; ++mt)
#pragma unroll
    for (int c = 0; c < KC; ++c)
      A[mt][c] = load_fragA_lds(sIn + mt * 16 * (KC * 32), lane, c * 32, KC * 32);
#pragma unroll
  for (int t = 0; t < 8; ++t) {   // hidden = 128 = 8 N-tiles
    v8f acc[MT] = {};
#pragma unroll
    for (int c = 0; c < KC; ++c) {
      v16h bf = load_fragB(pW1 + (size_t)(c * 8 + t) * 512, lane);
#pragma unroll
      for (int mt = 0; mt < MT; ++mt) acc[mt] = WMMA_F16(A[mt][c], bf, acc[mt]);
    }
    float bias = b1[t * 16 + nl];
#pragma unroll
    for (int mt = 0; mt < MT; ++mt)
#pragma unroll
      for (int v = 0; v < 8; ++v)
        sH[(mt * 16 + v + mh) * 128 + t * 16 + nl] = (_Float16)fmaxf(acc[mt][v] + bias, 0.0f);
  }
  __syncthreads();
  v16h A2[MT][4];
#pragma unroll
  for (int mt = 0; mt < MT; ++mt)
#pragma unroll
    for (int c = 0; c < 4; ++c)
      A2[mt][c] = load_fragA_lds(sH + mt * 16 * 128, lane, c * 32, 128);
#pragma unroll
  for (int t = 0; t < 4; ++t) {   // out = 64 = 4 N-tiles
    v8f acc[MT] = {};
#pragma unroll
    for (int c = 0; c < 4; ++c) {
      v16h bf = load_fragB(pW2 + (size_t)(c * 4 + t) * 512, lane);
#pragma unroll
      for (int mt = 0; mt < MT; ++mt) acc[mt] = WMMA_F16(A2[mt][c], bf, acc[mt]);
    }
    float bias = b2[t * 16 + nl];
#pragma unroll
    for (int mt = 0; mt < MT; ++mt) {
#pragma unroll
      for (int v = 0; v < 8; ++v) acc[mt][v] += bias;
      acc2[mt][t] = acc[mt];
    }
  }
}

// ---- prep: pack f32 weight [K,N] into f16 B-fragment blocks ----
__global__ void kPackW(const float* __restrict__ src, _Float16* __restrict__ dst,
                       int Kdim, int Ndim) {
  int total = (Kdim / 32) * (Ndim / 16) * 512;
  for (int t = blockIdx.x * blockDim.x + threadIdx.x; t < total; t += gridDim.x * blockDim.x) {
    int idx = t >> 9;          // fragment index = c*(N/16)+tt
    int w = t & 511;
    int lane = w >> 4, h = w & 15;
    int nt = Ndim / 16;
    int c = idx / nt, tt = idx % nt;
    int k = c * 32 + kmapB(h, lane);
    int n = tt * 16 + (lane & 15);
    dst[t] = (_Float16)src[(size_t)k * Ndim + n];
  }
}

// ---- prep: 2->64 encoder: out = relu(in @ W + b), optional clean(1/x) on input ----
__global__ void kEncode(const float* __restrict__ in, const float* __restrict__ W,
                        const float* __restrict__ b, _Float16* __restrict__ out,
                        int N, int invert) {
  int t = blockIdx.x * blockDim.x + threadIdx.x;
  if (t >= N * 64) return;
  int n = t >> 6, j = t & 63;
  float x0 = in[(size_t)n * 2], x1 = in[(size_t)n * 2 + 1];
  if (invert) {
    x0 = 1.0f / x0; if (!__builtin_isfinite(x0)) x0 = 0.0f;
    x1 = 1.0f / x1; if (!__builtin_isfinite(x1)) x1 = 0.0f;
  }
  float h = x0 * W[j] + x1 * W[64 + j] + b[j];
  out[t] = (_Float16)fmaxf(h, 0.0f);
}

// ---- prep: segsum[n,2] = seg(spacing,row) + seg(spacing,col) ----
__global__ void kSegSpacing(const int* __restrict__ rowi, const int* __restrict__ coli,
                            const float* __restrict__ edge_attr, float* __restrict__ segsum,
                            int BE) {
  int e = blockIdx.x * blockDim.x + threadIdx.x;
  if (e >= BE) return;
  float s0 = edge_attr[(size_t)e * 2], s1 = edge_attr[(size_t)e * 2 + 1];
  atomicAdd(&segsum[(size_t)rowi[e] * 2 + 0], s0);
  atomicAdd(&segsum[(size_t)rowi[e] * 2 + 1], s1);
  atomicAdd(&segsum[(size_t)coli[e] * 2 + 0], s0);
  atomicAdd(&segsum[(size_t)coli[e] * 2 + 1], s1);
}

// ---- edge kernel 1: d_edge = mlp([node_emb[row], node_emb[col], inv_spacing_emb]) ----
// one wave per 32-edge tile (MT=2); scatter-add into hj_plus[row], hj_minus[col].
__global__ __launch_bounds__(32)
void kDEdge(const int* __restrict__ rowi, const int* __restrict__ coli,
            const float* __restrict__ edge_attr,
            const float* __restrict__ enc2_W, const float* __restrict__ enc2_b,
            const _Float16* __restrict__ node_emb,
            const _Float16* __restrict__ pW1, const float* __restrict__ b1,
            const _Float16* __restrict__ pW2, const float* __restrict__ b2,
            float* __restrict__ hj_plus, float* __restrict__ hj_minus) {
  __shared__ __align__(16) _Float16 sIn[32 * 192];
  __shared__ __align__(16) _Float16 sH[32 * 128];
  __shared__ int sRow[32], sCol[32];
  const int lane = threadIdx.x;
  const int e0 = blockIdx.x * 32;
  sRow[lane] = rowi[e0 + lane];
  sCol[lane] = coli[e0 + lane];
  __syncthreads();
  // gather node_emb rows as 16B chunks
  for (int i = lane; i < 32 * 8; i += 32) {
    int m = i >> 3, c = i & 7;
    *(uint4*)&sIn[m * 192 + c * 8] = *(const uint4*)(node_emb + (size_t)sRow[m] * 64 + c * 8);
  }
  for (int i = lane; i < 32 * 8; i += 32) {
    int m = i >> 3, c = i & 7;
    *(uint4*)&sIn[m * 192 + 64 + c * 8] = *(const uint4*)(node_emb + (size_t)sCol[m] * 64 + c * 8);
  }
  // inv_spacing_emb computed on the fly (2->64 encoder, cleaned reciprocal)
  for (int i = lane; i < 32 * 64; i += 32) {
    int m = i >> 6, j = i & 63;
    int e = e0 + m;
    float s0 = edge_attr[(size_t)e * 2], s1 = edge_attr[(size_t)e * 2 + 1];
    float i0 = 1.0f / s0; if (!__builtin_isfinite(i0)) i0 = 0.0f;
    float i1 = 1.0f / s1; if (!__builtin_isfinite(i1)) i1 = 0.0f;
    sIn[m * 192 + 128 + j] =
        (_Float16)fmaxf(i0 * enc2_W[j] + i1 * enc2_W[64 + j] + enc2_b[j], 0.0f);
  }
  __syncthreads();
  v8f acc2[2][4];
  mlp2<6, 2>(sIn, sH, pW1, b1, pW2, b2, acc2, lane);
  const int mh = (lane & 16) ? 8 : 0, nl = lane & 15;
#pragma unroll
  for (int mt = 0; mt < 2; ++mt)
#pragma unroll
    for (int t = 0; t < 4; ++t)
#pragma unroll
      for (int v = 0; v < 8; ++v) {
        int m = mt * 16 + v + mh, n = t * 16 + nl;
        float d = acc2[mt][t][v];
        atomicAdd(&hj_plus[(size_t)sRow[m] * 64 + n], d);
        atomicAdd(&hj_minus[(size_t)sCol[m] * 64 + n], d);
      }
}

// ---- edge kernel 2: jacK = seg(mlp([spacing_emb, hj[idx]]), idx) (gather==scatter idx) ----
__global__ __launch_bounds__(32)
void kHEdge(const int* __restrict__ idx,
            const float* __restrict__ edge_attr,
            const float* __restrict__ encE_W, const float* __restrict__ encE_b,
            const float* __restrict__ hj,
            const _Float16* __restrict__ pW1, const float* __restrict__ b1,
            const _Float16* __restrict__ pW2, const float* __restrict__ b2,
            float* __restrict__ jacOut) {
  __shared__ __align__(16) _Float16 sIn[32 * 128];
  __shared__ __align__(16) _Float16 sH[32 * 128];
  __shared__ int sIdx[32];
  const int lane = threadIdx.x;
  const int e0 = blockIdx.x * 32;
  sIdx[lane] = idx[e0 + lane];
  __syncthreads();
  // spacing_emb computed on the fly (2->64 encoder)
  for (int i = lane; i < 32 * 64; i += 32) {
    int m = i >> 6, j = i & 63;
    int e = e0 + m;
    float s0 = edge_attr[(size_t)e * 2], s1 = edge_attr[(size_t)e * 2 + 1];
    sIn[m * 128 + j] = (_Float16)fmaxf(s0 * encE_W[j] + s1 * encE_W[64 + j] + encE_b[j], 0.0f);
  }
  // gather hj rows (f32) as float4 chunks, convert to f16
  for (int i = lane; i < 32 * 16; i += 32) {
    int m = i >> 4, c = i & 15;
    float4 f = *(const float4*)(hj + (size_t)sIdx[m] * 64 + c * 4);
    H4 h;
    h.v[0] = (_Float16)f.x; h.v[1] = (_Float16)f.y;
    h.v[2] = (_Float16)f.z; h.v[3] = (_Float16)f.w;
    *(H4*)&sIn[m * 128 + 64 + c * 4] = h;
  }
  __syncthreads();
  v8f acc2[2][4];
  mlp2<4, 2>(sIn, sH, pW1, b1, pW2, b2, acc2, lane);
  const int mh = (lane & 16) ? 8 : 0, nl = lane & 15;
#pragma unroll
  for (int mt = 0; mt < 2; ++mt)
#pragma unroll
    for (int t = 0; t < 4; ++t)
#pragma unroll
      for (int v = 0; v < 8; ++v)
        atomicAdd(&jacOut[(size_t)sIdx[mt * 16 + v + mh] * 64 + t * 16 + nl], acc2[mt][t][v]);
}

// ---- node kernel: jac-MLP, hess-MLP, res-MLP, decode; one wave per 16-node tile ----
__global__ __launch_bounds__(32)
void kNode(const int* __restrict__ batch, const float* __restrict__ node_attr,
           const _Float16* __restrict__ node_emb, const _Float16* __restrict__ invdbl_emb,
           const _Float16* __restrict__ glob_emb,
           const float* __restrict__ jac1, const float* __restrict__ jac2,
           const float* __restrict__ hjp, const float* __restrict__ hjm,
           const _Float16* __restrict__ pjW1, const float* __restrict__ jb1,
           const _Float16* __restrict__ pjW2, const float* __restrict__ jb2,
           const _Float16* __restrict__ psW1, const float* __restrict__ sb1,
           const _Float16* __restrict__ psW2, const float* __restrict__ sb2,
           const _Float16* __restrict__ pnW1, const float* __restrict__ nb1,
           const _Float16* __restrict__ pnW2, const float* __restrict__ nb2,
           const float* __restrict__ decW, const float* __restrict__ decb,
           float* __restrict__ out) {
  __shared__ __align__(16) _Float16 sIn[16 * 256];
  __shared__ __align__(16) _Float16 sH[16 * 128];
  __shared__ __align__(16) _Float16 sJac[16 * 64];
  __shared__ __align__(16) _Float16 sHess[16 * 64];
  __shared__ __align__(16) float sRes[16 * 64];
  __shared__ int sBatch[16];
  const int lane = threadIdx.x;
  const int n0 = blockIdx.x * 16;
  const int mh = (lane & 16) ? 8 : 0, nl = lane & 15;
  if (lane < 16) sBatch[lane] = batch[n0 + lane];
  v8f a2[1][4];

  // stage two f32 [*,64] sources + invdbl_emb into sIn[16x192] (vectorized)
  auto stage192 = [&](const float* __restrict__ f32a, const float* __restrict__ f32b) {
    for (int i = lane; i < 16 * 16; i += 32) {
      int m = i >> 4, c = i & 15;
      size_t nd = (size_t)(n0 + m);
      float4 fa = *(const float4*)(f32a + nd * 64 + c * 4);
      float4 fb = *(const float4*)(f32b + nd * 64 + c * 4);
      H4 ha, hb;
      ha.v[0] = (_Float16)fa.x; ha.v[1] = (_Float16)fa.y;
      ha.v[2] = (_Float16)fa.z; ha.v[3] = (_Float16)fa.w;
      hb.v[0] = (_Float16)fb.x; hb.v[1] = (_Float16)fb.y;
      hb.v[2] = (_Float16)fb.z; hb.v[3] = (_Float16)fb.w;
      *(H4*)&sIn[m * 192 + c * 4] = ha;
      *(H4*)&sIn[m * 192 + 64 + c * 4] = hb;
    }
    for (int i = lane; i < 16 * 8; i += 32) {
      int m = i >> 3, c = i & 7;
      *(uint4*)&sIn[m * 192 + 128 + c * 8] =
          *(const uint4*)(invdbl_emb + (size_t)(n0 + m) * 64 + c * 8);
    }
  };

  // --- jac = mlp([jac1, jac2, inv_double_emb]) ---
  stage192(jac1, jac2);
  __syncthreads();
  mlp2<6, 1>(sIn, sH, pjW1, jb1, pjW2, jb2, a2, lane);
  __syncthreads();
#pragma unroll
  for (int t = 0; t < 4; ++t)
#pragma unroll
    for (int v = 0; v < 8; ++v) sJac[(v + mh) * 64 + t * 16 + nl] = (_Float16)a2[0][t][v];
  __syncthreads();

  // --- hess = mlp([hj_plus, hj_minus, inv_double_emb]) ---
  stage192(hjp, hjm);
  __syncthreads();
  mlp2<6, 1>(sIn, sH, psW1, sb1, psW2, sb2, a2, lane);
  __syncthreads();
#pragma unroll
  for (int t = 0; t < 4; ++t)
#pragma unroll
    for (int v = 0; v < 8; ++v) sHess[(v + mh) * 64 + t * 16 + nl] = (_Float16)a2[0][t][v];
  __syncthreads();

  // --- res = mlp([node_emb, jac, glob_emb[batch], hess]) ---
  for (int i = lane; i < 16 * 8; i += 32) {
    int m = i >> 3, c = i & 7;
    size_t nd = (size_t)(n0 + m);
    *(uint4*)&sIn[m * 256 + c * 8] = *(const uint4*)(node_emb + nd * 64 + c * 8);
    *(uint4*)&sIn[m * 256 + 64 + c * 8] = *(const uint4*)&sJac[m * 64 + c * 8];
    *(uint4*)&sIn[m * 256 + 128 + c * 8] =
        *(const uint4*)(glob_emb + (size_t)sBatch[m] * 64 + c * 8);
    *(uint4*)&sIn[m * 256 + 192 + c * 8] = *(const uint4*)&sHess[m * 64 + c * 8];
  }
  __syncthreads();
  mlp2<8, 1>(sIn, sH, pnW1, nb1, pnW2, nb2, a2, lane);
#pragma unroll
  for (int t = 0; t < 4; ++t)
#pragma unroll
    for (int v = 0; v < 8; ++v) sRes[(v + mh) * 64 + t * 16 + nl] = a2[0][t][v];
  __syncthreads();

  // --- decode: out = node_attr * (res @ decW + decb); lane -> (node = lane&15, j = lane>>4)
  int m = lane & 15, j = lane >> 4;
  float s = decb[j];
#pragma unroll 8
  for (int f = 0; f < 64; ++f) s += sRes[m * 64 + f] * decW[f * 2 + j];
  size_t nd = (size_t)(n0 + m);
  out[nd * 2 + j] = node_attr[nd * 2 + j] * s;
}

// ------------------------------------------------------------------------------------
extern "C" void kernel_launch(void* const* d_in, const int* in_sizes, int n_in,
                              void* d_out, int out_size, void* d_ws, size_t ws_size,
                              hipStream_t stream) {
  const float* x          = (const float*)d_in[0];
  const int*   edge_index = (const int*)  d_in[1];
  const int*   batch      = (const int*)  d_in[2];
  const float* node_attr  = (const float*)d_in[3];
  const float* edge_attr  = (const float*)d_in[4];
  const float* glob_attr  = (const float*)d_in[5];
  const float* enc_node_W = (const float*)d_in[6];  const float* enc_node_b = (const float*)d_in[7];
  const float* enc_edge_W = (const float*)d_in[8];  const float* enc_edge_b = (const float*)d_in[9];
  const float* enc_glob_W = (const float*)d_in[10]; const float* enc_glob_b = (const float*)d_in[11];
  const float* enc2_W     = (const float*)d_in[12]; const float* enc2_b     = (const float*)d_in[13];
  const float* dW1 = (const float*)d_in[14]; const float* db1 = (const float*)d_in[15];
  const float* dW2 = (const float*)d_in[16]; const float* db2 = (const float*)d_in[17];
  const float* hW1 = (const float*)d_in[18]; const float* hb1 = (const float*)d_in[19];
  const float* hW2 = (const float*)d_in[20]; const float* hb2 = (const float*)d_in[21];
  const float* jW1 = (const float*)d_in[22]; const float* jb1 = (const float*)d_in[23];
  const float* jW2 = (const float*)d_in[24]; const float* jb2 = (const float*)d_in[25];
  const float* sW1 = (const float*)d_in[26]; const float* sb1 = (const float*)d_in[27];
  const float* sW2 = (const float*)d_in[28]; const float* sb2 = (const float*)d_in[29];
  const float* nW1 = (const float*)d_in[30]; const float* nb1 = (const float*)d_in[31];
  const float* nW2 = (const float*)d_in[32]; const float* nb2 = (const float*)d_in[33];
  const float* decW = (const float*)d_in[34]; const float* decb = (const float*)d_in[35];

  const int BN = in_sizes[2];
  const int BE = in_sizes[1] / 2;
  const int B  = in_sizes[5] / 2;
  const int* rowi = edge_index;
  const int* coli = edge_index + BE;
  float* out = (float*)d_out;

  // ---- workspace carve ----
  char* base = (char*)d_ws;
  size_t off = 0;
  auto take = [&](size_t bytes) -> char* {
    char* p = base + off;
    off += (bytes + 255) & ~(size_t)255;
    return p;
  };
  _Float16* node_emb_h = (_Float16*)take((size_t)BN * 64 * 2);
  _Float16* invdbl_h   = (_Float16*)take((size_t)BN * 64 * 2);
  _Float16* glob_h     = (_Float16*)take((size_t)B * 64 * 2);
  float* segsum = (float*)take((size_t)BN * 2 * 4);
  float* hjp    = (float*)take((size_t)BN * 64 * 4);
  float* hjm    = (float*)take((size_t)BN * 64 * 4);
  float* jac1   = (float*)take((size_t)BN * 64 * 4);
  float* jac2   = (float*)take((size_t)BN * 64 * 4);
  // packed f16 weights (fragment-native): size = (K/32)*(N/16)*512 halves
  _Float16* pdW1 = (_Float16*)take(48 * 512 * 2);  // 192x128
  _Float16* pdW2 = (_Float16*)take(16 * 512 * 2);  // 128x64
  _Float16* phW1 = (_Float16*)take(32 * 512 * 2);  // 128x128
  _Float16* phW2 = (_Float16*)take(16 * 512 * 2);  // 128x64
  _Float16* pjW1 = (_Float16*)take(48 * 512 * 2);  // 192x128
  _Float16* pjW2 = (_Float16*)take(16 * 512 * 2);  // 128x64
  _Float16* psW1 = (_Float16*)take(48 * 512 * 2);  // 192x128
  _Float16* psW2 = (_Float16*)take(16 * 512 * 2);  // 128x64
  _Float16* pnW1 = (_Float16*)take(64 * 512 * 2);  // 256x128
  _Float16* pnW2 = (_Float16*)take(16 * 512 * 2);  // 128x64

  // ---- zero accumulators (deterministic every call) ----
  hipMemsetAsync(segsum, 0, (size_t)BN * 2 * 4, stream);
  hipMemsetAsync(hjp,    0, (size_t)BN * 64 * 4, stream);
  hipMemsetAsync(hjm,    0, (size_t)BN * 64 * 4, stream);
  hipMemsetAsync(jac1,   0, (size_t)BN * 64 * 4, stream);
  hipMemsetAsync(jac2,   0, (size_t)BN * 64 * 4, stream);

  // ---- pack weights to f16 fragments (stays hot in L2) ----
  auto pack = [&](const float* src, _Float16* dst, int K, int N) {
    int total = (K / 32) * (N / 16) * 512;
    kPackW<<<(total + 255) / 256, 256, 0, stream>>>(src, dst, K, N);
  };
  pack(dW1, pdW1, 192, 128); pack(dW2, pdW2, 128, 64);
  pack(hW1, phW1, 128, 128); pack(hW2, phW2, 128, 64);
  pack(jW1, pjW1, 192, 128); pack(jW2, pjW2, 128, 64);
  pack(sW1, psW1, 192, 128); pack(sW2, psW2, 128, 64);
  pack(nW1, pnW1, 256, 128); pack(nW2, pnW2, 128, 64);

  // ---- encoders ----
  kEncode<<<((size_t)BN * 64 + 255) / 256, 256, 0, stream>>>(x, enc_node_W, enc_node_b,
                                                             node_emb_h, BN, 0);
  kEncode<<<((size_t)B * 64 + 255) / 256, 256, 0, stream>>>(glob_attr, enc_glob_W, enc_glob_b,
                                                            glob_h, B, 0);
  kSegSpacing<<<(BE + 255) / 256, 256, 0, stream>>>(rowi, coli, edge_attr, segsum, BE);
  kEncode<<<((size_t)BN * 64 + 255) / 256, 256, 0, stream>>>(segsum, enc2_W, enc2_b,
                                                             invdbl_h, BN, 1);

  // ---- per-edge d-MLP + scatter into hj_plus / hj_minus (32 edges per wave) ----
  kDEdge<<<BE / 32, 32, 0, stream>>>(rowi, coli, edge_attr, enc2_W, enc2_b, node_emb_h,
                                     pdW1, db1, pdW2, db2, hjp, hjm);

  // ---- per-edge half-jacobian MLPs (gather index == scatter index) ----
  kHEdge<<<BE / 32, 32, 0, stream>>>(coli, edge_attr, enc_edge_W, enc_edge_b, hjp,
                                     phW1, hb1, phW2, hb2, jac1);
  kHEdge<<<BE / 32, 32, 0, stream>>>(rowi, edge_attr, enc_edge_W, enc_edge_b, hjm,
                                     phW1, hb1, phW2, hb2, jac2);

  // ---- per-node jac / hess / res MLPs + decode ----
  kNode<<<BN / 16, 32, 0, stream>>>(batch, node_attr, node_emb_h, invdbl_h, glob_h,
                                    jac1, jac2, hjp, hjm,
                                    pjW1, jb1, pjW2, jb2,
                                    psW1, sb1, psW2, sb2,
                                    pnW1, nb1, pnW2, nb2,
                                    decW, decb, out);
}
